// SwinTransformer_82222853915239
// MI455X (gfx1250) — matrix operand order repaired and checked by reading, not compiled, not run on hardware
//
#include <hip/hip_runtime.h>
#include <hip/hip_bf16.h>
#include <math.h>

// ---------------------------------------------------------------------------
// Swin Transformer block on gfx1250 (MI455X): all GEMMs via
// v_wmma_f32_16x16x32_f16 with LDS-staged block tiles (async global->LDS when
// available), LDS-staged attention, fused LN/shift/softmax/GELU epilogues.
// ~163 GFLOP total, ~0.6 GB HBM traffic -> balanced at 23.3 TB/s.
// ---------------------------------------------------------------------------

typedef __attribute__((ext_vector_type(16))) _Float16 v16h;
typedef __attribute__((ext_vector_type(8)))  float    v8f;
typedef __attribute__((ext_vector_type(4)))  int      v4i;

#define B_SZ   32
#define HH     56
#define WW     56
#define CC     256
#define NHEAD  8
#define WS7    7
#define NTOK   49          // tokens per window
#define NWIN   64          // windows per image
#define HD     32          // head dim
#define HID    1024
#define MROWS  (B_SZ * NWIN * NTOK)   // 100352 rows (both window & token order)

// ---- WMMA helpers ----------------------------------------------------------

__device__ __forceinline__ v8f wmma32(v16h a, v16h b, v8f c) {
  return __builtin_amdgcn_wmma_f32_16x16x32_f16(
      /*neg_a=*/false, a, /*neg_b=*/false, b,
      /*c_mod=*/(short)0, c, /*reuse_a=*/false, /*reuse_b=*/false);
}

// A fragment (16x32, row-major source, ld elems): lane<16 -> row m0+lane,
// halves K = [kc..kc+7, kc+16..kc+23]; lane>=16 -> same row, K offset +8.
template <typename PT>
__device__ __forceinline__ v16h load_a16(PT A, int ld, int m0, int kc) {
  int l    = threadIdx.x & 31;
  long row = m0 + (l & 15);
  int  kof = kc + ((l >> 4) << 3);
  v16h r;
#pragma unroll
  for (int i = 0; i < 8; ++i) r[i]     = A[row * ld + kof + i];
#pragma unroll
  for (int i = 0; i < 8; ++i) r[8 + i] = A[row * ld + kof + 16 + i];
  return r;
}

// B fragment (32x16 of Wt[K,N], row-major, ld elems): lane<16 -> col n0+lane,
// K = [kc..kc+7, kc+16..kc+23]; lane>=16 -> same col, K offset +8.
template <typename PT>
__device__ __forceinline__ v16h load_b16(PT Bm, int ld, int kc, int n0) {
  int l    = threadIdx.x & 31;
  long col = n0 + (l & 15);
  int  kof = kc + ((l >> 4) << 3);
  v16h r;
#pragma unroll
  for (int i = 0; i < 8; ++i) r[i]     = Bm[(long)(kof + i) * ld + col];
#pragma unroll
  for (int i = 0; i < 8; ++i) r[8 + i] = Bm[(long)(kof + 16 + i) * ld + col];
  return r;
}

// ---- 16B global->LDS copy: async (GLOBAL_LOAD_ASYNC_TO_LDS_B128) if the
// toolchain exposes it, else a plain vector copy through VGPRs. ------------

#if __has_builtin(__builtin_amdgcn_global_load_async_to_lds_b128)
#define HAVE_ASYNC_LDS 1
#else
#define HAVE_ASYNC_LDS 0
#endif

typedef __attribute__((address_space(1))) v4i* gptr_v4i;
typedef __attribute__((address_space(3))) v4i* lptr_v4i;

__device__ __forceinline__ void cp16(_Float16* dst_lds, const _Float16* src_g) {
#if HAVE_ASYNC_LDS
  v4i* sg = (v4i*)src_g;    // drop const, retype (generic)
  v4i* dl = (v4i*)dst_lds;
  __builtin_amdgcn_global_load_async_to_lds_b128(
      (gptr_v4i)sg, (lptr_v4i)dl, 0, 0);
#else
  *(uint4*)dst_lds = *(const uint4*)src_g;
#endif
}

__device__ __forceinline__ void async_fence() {
#if HAVE_ASYNC_LDS
#if __has_builtin(__builtin_amdgcn_s_wait_asynccnt)
  __builtin_amdgcn_s_wait_asynccnt(0);
#else
  asm volatile("s_wait_asynccnt 0" ::: "memory");
#endif
#endif
}

// ---- weight transpose + f32->f16 ------------------------------------------

__global__ void transpose_w(const float* __restrict__ W, _Float16* __restrict__ Wt,
                            int Nout, int K) {
  long i = (long)blockIdx.x * 256 + threadIdx.x;
  if (i < (long)Nout * K) {
    int n = (int)(i / K), k = (int)(i % K);
    Wt[(long)k * Nout + n] = (_Float16)W[i];
  }
}

// ---- LayerNorm (optionally fused with shift + window partition) ------------
__global__ __launch_bounds__(256) void ln_kernel(const float* __restrict__ x,
                                                 const float* __restrict__ g,
                                                 const float* __restrict__ bt,
                                                 _Float16* __restrict__ out,
                                                 int windowed) {
  int  wave = threadIdx.x >> 5, lane = threadIdx.x & 31;
  long r    = (long)blockIdx.x * 8 + wave;
  long src  = r;
  if (windowed) {
    int b_  = (int)(r / NTOK), nt = (int)(r % NTOK);
    int wix = b_ & (NWIN - 1), b = b_ >> 6;
    int wr = wix >> 3, wc = wix & 7;
    int lr = nt / WS7, lc = nt - lr * WS7;
    int h0 = wr * WS7 + lr + 3; if (h0 >= HH) h0 -= HH;
    int w0 = wc * WS7 + lc + 3; if (w0 >= WW) w0 -= WW;
    src = (long)b * (HH * WW) + h0 * WW + w0;
  }
  const float* xr = x + src * CC;
  int c0 = lane * 8;
  float v[8], s = 0.f, s2 = 0.f;
#pragma unroll
  for (int i = 0; i < 8; ++i) { v[i] = xr[c0 + i]; s += v[i]; s2 += v[i] * v[i]; }
#pragma unroll
  for (int off = 16; off > 0; off >>= 1) {
    s  += __shfl_xor(s,  off, 32);
    s2 += __shfl_xor(s2, off, 32);
  }
  float mu = s * (1.f / CC);
  float rs = rsqrtf(fmaxf(s2 * (1.f / CC) - mu * mu, 0.f) + 1e-5f);
#pragma unroll
  for (int i = 0; i < 8; ++i)
    out[r * CC + c0 + i] = (_Float16)((v[i] - mu) * rs * g[c0 + i] + bt[c0 + i]);
}

// ---- generic WMMA GEMM, LDS-staged block tiles, fused epilogues ------------
// C[M,Nout] = A[M,K](f16) x Wt[K,Nout](f16) + bias. Block tile 128x128,
// 8 waves, wave tile 32x64. Per 32-K chunk: stage A(128x32) and Wt(32x128)
// into LDS (async global->LDS b128 when available), then 8 wmma per wave from
// ds_load_b128 fragments. All dims divide tiles exactly (EXEC all-ones).
// MODE 0: store f16 (qkv).           MODE 1: +bias, reverse shift scatter,
// MODE 2: GELU(exact erf) -> f16.            +residual(extra) -> f32 (proj).
// MODE 3: +bias +extra -> f32 (fc2 -> d_out).
template <int MODE>
__global__ __launch_bounds__(256) void gemm_ep(const _Float16* __restrict__ A,
                                               const _Float16* __restrict__ Wt,
                                               const float* __restrict__ bias,
                                               void* __restrict__ outp,
                                               const float* __restrict__ extra,
                                               int Nout, int K) {
  __shared__ _Float16 sA[128 * 32];   // 8 KB
  __shared__ _Float16 sB[32 * 128];   // 8 KB

  int tid  = threadIdx.x;
  int wave = tid >> 5, lane = tid & 31;
  int bm = blockIdx.y * 128, bn = blockIdx.x * 128;
  int m0 = (wave & 3) * 32;   // local row within block tile
  int n0 = (wave >> 2) * 64;  // local col within block tile

  v8f zero = {0.f, 0.f, 0.f, 0.f, 0.f, 0.f, 0.f, 0.f};
  v8f acc[2][4];
#pragma unroll
  for (int i = 0; i < 2; ++i)
#pragma unroll
    for (int j = 0; j < 4; ++j) acc[i][j] = zero;

  for (int kc = 0; kc < K; kc += 32) {
    // ---- stage A(128x32) and B(32x128) chunks into LDS (1024 x 16B total) --
#pragma unroll
    for (int q = 0; q < 2; ++q) {
      int e  = tid + q * 256;
      int ra = e >> 2, ca = (e & 3) * 8;            // A: 4 x 16B per row
      cp16(&sA[ra * 32 + ca], A + (long)(bm + ra) * K + kc + ca);
      int rb = e >> 4, cb = (e & 15) * 8;           // B: 16 x 16B per row
      cp16(&sB[rb * 128 + cb], Wt + (long)(kc + rb) * Nout + bn + cb);
    }
    async_fence();
    __syncthreads();

    v16h a0 = load_a16(sA, 32, m0, 0);
    v16h a1 = load_a16(sA, 32, m0 + 16, 0);
#pragma unroll
    for (int j = 0; j < 4; ++j) {
      v16h b = load_b16(sB, 128, 0, n0 + j * 16);
      acc[0][j] = wmma32(a0, b, acc[0][j]);
      acc[1][j] = wmma32(a1, b, acc[1][j]);
    }
    __syncthreads();  // protect LDS before next-chunk staging
  }

  int ln = lane & 15;
  int mh = (lane >> 4) * 8;
#pragma unroll
  for (int i = 0; i < 2; ++i) {
#pragma unroll
    for (int j = 0; j < 4; ++j) {
      int n = bn + n0 + j * 16 + ln;
      float bnv = bias[n];
#pragma unroll
      for (int r = 0; r < 8; ++r) {
        int m = bm + m0 + i * 16 + mh + r;
        float val = acc[i][j][r] + bnv;
        if (MODE == 0) {
          ((_Float16*)outp)[(long)m * Nout + n] = (_Float16)val;
        } else if (MODE == 1) {
          int b_ = m / NTOK, nt = m - b_ * NTOK;
          int wix = b_ & (NWIN - 1), b = b_ >> 6;
          int wr = wix >> 3, wc = wix & 7;
          int lr = nt / WS7, lc = nt - lr * WS7;
          int h0 = wr * WS7 + lr + 3; if (h0 >= HH) h0 -= HH;
          int w0 = wc * WS7 + lc + 3; if (w0 >= WW) w0 -= WW;
          long row = (long)b * (HH * WW) + h0 * WW + w0;
          ((float*)outp)[row * CC + n] = extra[row * CC + n] + val;
        } else if (MODE == 2) {
          float gel = 0.5f * val * (1.f + erff(val * 0.70710678118654752f));
          ((_Float16*)outp)[(long)m * Nout + n] = (_Float16)gel;
        } else {
          long idx = (long)m * Nout + n;
          ((float*)outp)[idx] = extra[idx] + val;
        }
      }
    }
  }
}

// ---- windowed attention: one block per (window, head) ----------------------
__global__ __launch_bounds__(128) void attn_kernel(const _Float16* __restrict__ qkv,
                                                   const float* __restrict__ rpb,
                                                   const int* __restrict__ rel_idx,
                                                   const float* __restrict__ mask,
                                                   _Float16* __restrict__ o) {
  __shared__ _Float16 qs[64 * 32];   // Q, pre-scaled, rows padded to 64
  __shared__ _Float16 kts[32 * 64];  // K^T (HD x tokens)
  __shared__ _Float16 vs[64 * 32];   // V
  __shared__ float    sc[64 * 64];   // scores f32
  __shared__ _Float16 ps[64 * 64];   // softmax probs f16

  int bh = blockIdx.x;
  int b_ = bh >> 3, h = bh & 7;
  int wix = b_ & (NWIN - 1);
  int tid = threadIdx.x;
  const float scale = 0.17677669529663687f;  // HD^-0.5

  const _Float16* base = qkv + (long)b_ * NTOK * (3 * CC) + h * HD;
  for (int i = tid; i < 64 * 32; i += 128) {
    int n = i >> 5, d = i & 31;
    _Float16 qv = (_Float16)0.f, kv = (_Float16)0.f, vv = (_Float16)0.f;
    if (n < NTOK) {
      const _Float16* row = base + (long)n * (3 * CC);
      qv = (_Float16)((float)row[d] * scale);
      kv = row[CC + d];
      vv = row[2 * CC + d];
    }
    qs[n * 32 + d] = qv;
    kts[d * 64 + n] = kv;
    vs[n * 32 + d] = vv;
  }
  __syncthreads();

  int wave = tid >> 5, lane = tid & 31;
  int m0 = wave * 16;
  int ln = lane & 15, mh = (lane >> 4) * 8;

  // scores = Q x K^T  (4 N-tiles per wave, K-dim = 32 -> one wmma each)
  {
    v16h a = load_a16(qs, 32, m0, 0);
#pragma unroll
    for (int j = 0; j < 4; ++j) {
      v16h b = load_b16(kts, 64, 0, j * 16);
      v8f c = {0.f, 0.f, 0.f, 0.f, 0.f, 0.f, 0.f, 0.f};
      c = wmma32(a, b, c);
      int n = j * 16 + ln;
#pragma unroll
      for (int r = 0; r < 8; ++r) sc[(m0 + mh + r) * 64 + n] = c[r];
    }
  }
  __syncthreads();

  // per-row softmax with relative-position bias + shift mask
  if (tid < 64) {
    int m = tid;
    if (m < NTOK) {
      float mx = -1e30f;
      for (int n = 0; n < NTOK; ++n) {
        float s = sc[m * 64 + n]
                + rpb[rel_idx[m * NTOK + n] * NHEAD + h]
                + mask[(long)wix * (NTOK * NTOK) + m * NTOK + n];
        sc[m * 64 + n] = s;
        mx = fmaxf(mx, s);
      }
      float sum = 0.f;
      for (int n = 0; n < NTOK; ++n) {
        float e = __expf(sc[m * 64 + n] - mx);
        sc[m * 64 + n] = e;
        sum += e;
      }
      float inv = 1.f / sum;
      for (int n = 0; n < 64; ++n)
        ps[m * 64 + n] = (n < NTOK) ? (_Float16)(sc[m * 64 + n] * inv)
                                    : (_Float16)0.f;
    } else {
      for (int n = 0; n < 64; ++n) ps[m * 64 + n] = (_Float16)0.f;
    }
  }
  __syncthreads();

  // O = P x V  (2 N-tiles of HD, K-dim = 64 -> 2 wmma each)
#pragma unroll
  for (int j = 0; j < 2; ++j) {
    v8f c = {0.f, 0.f, 0.f, 0.f, 0.f, 0.f, 0.f, 0.f};
#pragma unroll
    for (int kc = 0; kc < 64; kc += 32) {
      v16h a = load_a16(ps, 64, m0, kc);
      v16h b = load_b16(vs, 32, kc, j * 16);
      c = wmma32(a, b, c);
    }
    int n = j * 16 + ln;
#pragma unroll
    for (int r = 0; r < 8; ++r) {
      int m = m0 + mh + r;
      if (m < NTOK)
        o[((long)b_ * NTOK + m) * CC + h * HD + n] = (_Float16)c[r];
    }
  }
}

// ---------------------------------------------------------------------------

extern "C" void kernel_launch(void* const* d_in, const int* in_sizes, int n_in,
                              void* d_out, int out_size, void* d_ws, size_t ws_size,
                              hipStream_t stream) {
  (void)in_sizes; (void)n_in; (void)out_size; (void)ws_size;

  const float* x       = (const float*)d_in[0];
  const float* norm1_g = (const float*)d_in[1];
  const float* norm1_b = (const float*)d_in[2];
  const float* qkv_w   = (const float*)d_in[3];
  const float* qkv_b   = (const float*)d_in[4];
  const float* rpb     = (const float*)d_in[5];
  const float* proj_w  = (const float*)d_in[6];
  const float* proj_b  = (const float*)d_in[7];
  const float* norm2_g = (const float*)d_in[8];
  const float* norm2_b = (const float*)d_in[9];
  const float* fc1_w   = (const float*)d_in[10];
  const float* fc1_b   = (const float*)d_in[11];
  const float* fc2_w   = (const float*)d_in[12];
  const float* fc2_b   = (const float*)d_in[13];
  const float* mask    = (const float*)d_in[14];
  const int*   rel_idx = (const int*)d_in[15];

  char* ws = (char*)d_ws;
  const long SZ_H = (long)MROWS * CC * 2;            // 51,380,224 B
  _Float16* xw      = (_Float16*)(ws);
  _Float16* qkvb    = (_Float16*)(ws + SZ_H);                       // M x 768 f16
  _Float16* obuf    = (_Float16*)(ws + SZ_H + 3 * SZ_H);            // M x 256 f16
  float*    x2      = (float*)   (ws + 5 * SZ_H);                   // M x 256 f32
  _Float16* h2      = (_Float16*)(ws + 7 * SZ_H);                   // M x 256 f16
  _Float16* act     = (_Float16*)(ws + 8 * SZ_H);                   // M x 1024 f16
  _Float16* qkv_wt  = (_Float16*)(ws + 12 * SZ_H);
  _Float16* proj_wt = qkv_wt + 768 * 256;
  _Float16* fc1_wt  = proj_wt + 256 * 256;
  _Float16* fc2_wt  = fc1_wt + 1024 * 256;

  // weight transpose/convert (tiny)
  transpose_w<<<768,  256, 0, stream>>>(qkv_w,  qkv_wt,  768,  256);
  transpose_w<<<256,  256, 0, stream>>>(proj_w, proj_wt, 256,  256);
  transpose_w<<<1024, 256, 0, stream>>>(fc1_w,  fc1_wt,  1024, 256);
  transpose_w<<<1024, 256, 0, stream>>>(fc2_w,  fc2_wt,  256,  1024);

  // LN1 + cyclic shift + window partition -> xw (window-token order, f16)
  ln_kernel<<<MROWS / 8, 256, 0, stream>>>(x, norm1_g, norm1_b, xw, 1);

  // QKV GEMM: 100352 x 768 x 256
  gemm_ep<0><<<dim3(768 / 128, MROWS / 128), 256, 0, stream>>>(
      xw, qkv_wt, qkv_b, qkvb, nullptr, 768, 256);

  // windowed attention per (window, head)
  attn_kernel<<<B_SZ * NWIN * NHEAD, 128, 0, stream>>>(qkvb, rpb, rel_idx, mask, obuf);

  // proj GEMM + reverse shift scatter + residual -> x2 (f32)
  gemm_ep<1><<<dim3(256 / 128, MROWS / 128), 256, 0, stream>>>(
      obuf, proj_wt, proj_b, x2, x, 256, 256);

  // LN2 -> h2 (f16)
  ln_kernel<<<MROWS / 8, 256, 0, stream>>>(x2, norm2_g, norm2_b, h2, 0);

  // FC1 + GELU -> act (f16)
  gemm_ep<2><<<dim3(1024 / 128, MROWS / 128), 256, 0, stream>>>(
      h2, fc1_wt, fc1_b, act, nullptr, 1024, 256);

  // FC2 + residual -> d_out (f32)
  gemm_ep<3><<<dim3(256 / 128, MROWS / 128), 256, 0, stream>>>(
      act, fc2_wt, fc2_b, d_out, x2, 256, 1024);
}
